// LoraLinear_82360292868555
// MI455X (gfx1250) — matrix operand order repaired and checked by reading, not compiled, not run on hardware
//
#include <hip/hip_runtime.h>
#include <hip/hip_bf16.h>

typedef __attribute__((ext_vector_type(8)))  __bf16 v8bf;
typedef __attribute__((ext_vector_type(16))) __bf16 v16bf;
typedef __attribute__((ext_vector_type(8)))  float  v8f;

// Problem constants
#define IN_DIM   4096
#define OUT_DIM  11008
#define R_DIM    64
#define M_DIM    256      // B*S = 8*32
#define SCALING  2.0f
#define EPSV     1e-6f

// ---------------------------------------------------------------------------
// CDNA5 async global->LDS helpers (gfx1250).  The generic address of a
// __shared__ object has the LDS byte offset in its low 32 bits (ISA 10.2).
// ---------------------------------------------------------------------------
__device__ __forceinline__ unsigned lds_off(const void* p) {
    return (unsigned)(unsigned long long)p;
}
__device__ __forceinline__ void async_load_b128(unsigned ldsOff, const void* gaddr) {
    asm volatile("global_load_async_to_lds_b128 %0, %1, off"
                 :: "v"(ldsOff), "v"(gaddr) : "memory");
}
__device__ __forceinline__ void wait_asynccnt0() {
    asm volatile("s_wait_asynccnt 0x0" ::: "memory");
}

// ---------------------------------------------------------------------------
// 1) zero the per-column norm accumulator
// ---------------------------------------------------------------------------
__global__ __launch_bounds__(256) void zero_kernel(float* __restrict__ acc) {
    int i = blockIdx.x * 256 + threadIdx.x;
    if (i < IN_DIM) acc[i] = 0.0f;
}

// ---------------------------------------------------------------------------
// 2) per-IN-column sum of squares of (W + 2*B@A), computed on the fly.
//    grid (IN/256, OUT/256). Thread owns one column i; A[:,i] lives in 64 VGPRs.
// ---------------------------------------------------------------------------
__global__ __launch_bounds__(256) void norm_kernel(
    const float* __restrict__ weight,   // [OUT, IN]
    const float* __restrict__ loraA,    // [R, IN]
    const float* __restrict__ loraB,    // [OUT, R]
    float* __restrict__ normAcc)        // [IN]
{
    constexpr int CH = 64;              // B rows staged per chunk
    constexpr int ROWS = 256;           // rows handled per block
    __shared__ float sB[CH * R_DIM];    // 16 KB

    const int tid = threadIdx.x;
    const int i   = blockIdx.x * 256 + tid;   // column index
    const int o0  = blockIdx.y * ROWS;

    float areg[R_DIM];
    #pragma unroll
    for (int r = 0; r < R_DIM; ++r) areg[r] = loraA[(long)r * IN_DIM + i];

    float ss = 0.0f;
    for (int oc = 0; oc < ROWS / CH; ++oc) {
        const int ob = o0 + oc * CH;
        __syncthreads();
        #pragma unroll
        for (int j = 0; j < (CH * R_DIM) / 256; ++j) {   // 16 each
            int idx = tid + j * 256;
            sB[idx] = loraB[(long)ob * R_DIM + idx];
        }
        __syncthreads();
        for (int r2 = 0; r2 < CH; ++r2) {
            float lw = 0.0f;
            #pragma unroll
            for (int q = 0; q < R_DIM; ++q) lw += sB[r2 * R_DIM + q] * areg[q];
            float wv = weight[(long)(ob + r2) * IN_DIM + i];
            float v  = wv + SCALING * lw;
            ss += v * v;
        }
    }
    atomicAdd(&normAcc[i], ss);
}

// ---------------------------------------------------------------------------
// 3) scale[i] = m[i] / (sqrt(ss[i]) + eps)
// ---------------------------------------------------------------------------
__global__ __launch_bounds__(256) void scale_kernel(
    const float* __restrict__ acc, const float* __restrict__ loraM,
    float* __restrict__ scalev)
{
    int i = blockIdx.x * 256 + threadIdx.x;
    if (i < IN_DIM) scalev[i] = loraM[i] / (sqrtf(acc[i]) + EPSV);
}

// ---------------------------------------------------------------------------
// 4) x_hat (bf16) = x * scale  (DoRA column scale folded into activations)
// ---------------------------------------------------------------------------
__global__ __launch_bounds__(256) void xs_kernel(
    const float* __restrict__ x, const float* __restrict__ scalev,
    __bf16* __restrict__ xs)
{
    int idx = blockIdx.x * 256 + threadIdx.x;   // < M_DIM*IN_DIM
    int k = idx & (IN_DIM - 1);
    xs[idx] = (__bf16)(x[idx] * scalev[k]);
}

// ---------------------------------------------------------------------------
// 5) t_aug[m,r] = 2 * sum_k x_hat[m,k] * A[r,k]   (bf16, [256 x 64])
// ---------------------------------------------------------------------------
__global__ __launch_bounds__(64) void t_kernel(
    const __bf16* __restrict__ xs, const float* __restrict__ loraA,
    __bf16* __restrict__ taug)
{
    int m = blockIdx.x, r = threadIdx.x;
    const __bf16* xrow = xs + (long)m * IN_DIM;
    const float*  arow = loraA + (long)r * IN_DIM;
    float t = 0.0f;
    for (int k = 0; k < IN_DIM; ++k) t += (float)xrow[k] * arow[k];
    taug[m * R_DIM + r] = (__bf16)(SCALING * t);
}

// ---------------------------------------------------------------------------
// 6) Main GEMM:  out[m,n] = sum_k x_hat[m,k]*W[n,k] + sum_r t_aug[m,r]*B[n,r] + bias[n]
//    128x128 block tile, 8 wave32 waves, K-step 32, bf16 WMMA, f32 accumulate.
//    Double-buffered LDS: next x tile staged via global_load_async_to_lds_b128,
//    next W tile prefetched into registers (needs fp32->bf16 convert) while the
//    current tile's WMMAs run.  LoRA correction is a separate 2-step tail.
// ---------------------------------------------------------------------------
__global__ __launch_bounds__(256) void gemm_kernel(
    const float*  __restrict__ weight,  // [OUT, IN] fp32
    const float*  __restrict__ loraB,   // [OUT, R]  fp32
    const __bf16* __restrict__ xs,      // [M, IN]   bf16 (scaled)
    const __bf16* __restrict__ taug,    // [M, R]    bf16 (2*x_hat@A^T)
    const float*  __restrict__ bias,    // [OUT]
    float*        __restrict__ out)     // [M, OUT]
{
    constexpr int BM = 128, BN = 128, BK = 32;
    constexpr int KSTEPS = IN_DIM / BK;          // 128

    __shared__ __bf16 sX[2][BM * BK];   // 2 x 8 KB
    __shared__ __bf16 sW[2][BN * BK];   // 2 x 8 KB

    const int tid   = threadIdx.x;
    const int lane  = tid & 31;
    const int wid   = tid >> 5;          // 0..7
    const int wm    = wid & 3;           // 4 M-groups of 32 rows
    const int wn    = wid >> 2;          // 2 N-groups of 64 cols
    const int lrow  = lane & 15;
    const int lhalf = lane >> 4;
    const int nBase = blockIdx.x * BN;
    const int mBase = blockIdx.y * BM;

    // staging coordinates
    const int wr   = tid >> 3;           // W: row 0..31 (+i*32), 4 rows/thread
    const int wc4  = (tid & 7) * 4;      // W: col 0..28 step 4
    const int xrow = tid >> 1;           // X: row 0..127
    const int xh   = (tid & 1) * 16;     // X: half-row offset 0/16

    const float*  wp = weight + (long)nBase * IN_DIM;
    const __bf16* xp = xs     + (long)mBase * IN_DIM;

    v8f acc[2][4];
    #pragma unroll
    for (int mi = 0; mi < 2; ++mi)
        #pragma unroll
        for (int ni = 0; ni < 4; ++ni)
            acc[mi][ni] = (v8f){0.f,0.f,0.f,0.f,0.f,0.f,0.f,0.f};

    // ---- prologue: stage k-step 0 into buffer 0 ----
    async_load_b128(lds_off(&sX[0][xrow * BK + xh]),
                    xp + (long)xrow * IN_DIM + xh);
    async_load_b128(lds_off(&sX[0][xrow * BK + xh + 8]),
                    xp + (long)xrow * IN_DIM + xh + 8);
    #pragma unroll
    for (int i = 0; i < 4; ++i) {
        const int row = wr + i * 32;
        const float4 v = *(const float4*)(wp + (long)row * IN_DIM + wc4);
        __bf16* d = &sW[0][row * BK + wc4];
        d[0] = (__bf16)v.x; d[1] = (__bf16)v.y;
        d[2] = (__bf16)v.z; d[3] = (__bf16)v.w;
    }
    wait_asynccnt0();
    __syncthreads();

    // ---- main K loop over the base weight ----
    for (int kk = 0; kk < KSTEPS; ++kk) {
        const int cur = kk & 1, nxt = cur ^ 1;
        const bool hasNext = (kk + 1 < KSTEPS);
        float4 wreg[4];

        if (hasNext) {
            const int k0 = (kk + 1) * BK;
            // next x tile: async straight into LDS (already bf16)
            async_load_b128(lds_off(&sX[nxt][xrow * BK + xh]),
                            xp + (long)xrow * IN_DIM + k0 + xh);
            async_load_b128(lds_off(&sX[nxt][xrow * BK + xh + 8]),
                            xp + (long)xrow * IN_DIM + k0 + xh + 8);
            // next W tile: global -> registers (converted after compute)
            #pragma unroll
            for (int i = 0; i < 4; ++i)
                wreg[i] = *(const float4*)(wp + (long)(wr + i * 32) * IN_DIM + k0 + wc4);
        }

        // ---- compute current tile (ISA 7.12.2 fragment layouts, wave32) ----
        v16bf af[2], bfrag[4];
        #pragma unroll
        for (int mi = 0; mi < 2; ++mi) {
            // A 16x32 bf16: lane<16 holds K0..7 then K16..23; lane>=16 K8..15 / K24..31
            const __bf16* ap = &sX[cur][(wm * 32 + mi * 16 + lrow) * BK + lhalf * 8];
            v8bf lo = *(const v8bf*)ap;
            v8bf hi = *(const v8bf*)(ap + 16);
            af[mi] = __builtin_shufflevector(lo, hi, 0,1,2,3,4,5,6,7,8,9,10,11,12,13,14,15);
        }
        #pragma unroll
        for (int ni = 0; ni < 4; ++ni) {
            // B 32x16 bf16: lane<16 -> N=lane, K0..7 then K16..23; lane>=16 -> N=lane-16, K8..15 / K24..31
            const __bf16* bp = &sW[cur][(wn * 64 + ni * 16 + lrow) * BK + lhalf * 8];
            v8bf lo = *(const v8bf*)bp;
            v8bf hi = *(const v8bf*)(bp + 16);
            bfrag[ni] = __builtin_shufflevector(lo, hi, 0,1,2,3,4,5,6,7,8,9,10,11,12,13,14,15);
        }
        #pragma unroll
        for (int mi = 0; mi < 2; ++mi)
            #pragma unroll
            for (int ni = 0; ni < 4; ++ni)
                acc[mi][ni] = __builtin_amdgcn_wmma_f32_16x16x32_bf16(
                    false, af[mi], false, bfrag[ni],
                    (short)0, acc[mi][ni], false, false);

        if (hasNext) {
            // convert + store next W tile into the other buffer
            #pragma unroll
            for (int i = 0; i < 4; ++i) {
                __bf16* d = &sW[nxt][(wr + i * 32) * BK + wc4];
                d[0] = (__bf16)wreg[i].x; d[1] = (__bf16)wreg[i].y;
                d[2] = (__bf16)wreg[i].z; d[3] = (__bf16)wreg[i].w;
            }
            wait_asynccnt0();   // my async x stores to LDS have landed
        }
        __syncthreads();
    }

    // ---- LoRA tail: 2 K-steps over (taug [M x 64] bf16, lora_B [OUT x 64] f32) ----
    for (int ls = 0; ls < R_DIM / BK; ++ls) {
        const int k0 = ls * BK;
        #pragma unroll
        for (int i = 0; i < 4; ++i) {
            const int row = wr + i * 32;
            const float4 v = *(const float4*)(loraB + (long)(nBase + row) * R_DIM + k0 + wc4);
            __bf16* d = &sW[0][row * BK + wc4];
            d[0] = (__bf16)v.x; d[1] = (__bf16)v.y;
            d[2] = (__bf16)v.z; d[3] = (__bf16)v.w;
        }
        {
            const v8bf* s = (const v8bf*)(taug + (long)(mBase + xrow) * R_DIM + k0 + xh);
            *(v8bf*)&sX[0][xrow * BK + xh]     = s[0];
            *(v8bf*)&sX[0][xrow * BK + xh + 8] = s[1];
        }
        __syncthreads();

        v16bf af[2], bfrag[4];
        #pragma unroll
        for (int mi = 0; mi < 2; ++mi) {
            const __bf16* ap = &sX[0][(wm * 32 + mi * 16 + lrow) * BK + lhalf * 8];
            v8bf lo = *(const v8bf*)ap;
            v8bf hi = *(const v8bf*)(ap + 16);
            af[mi] = __builtin_shufflevector(lo, hi, 0,1,2,3,4,5,6,7,8,9,10,11,12,13,14,15);
        }
        #pragma unroll
        for (int ni = 0; ni < 4; ++ni) {
            const __bf16* bp = &sW[0][(wn * 64 + ni * 16 + lrow) * BK + lhalf * 8];
            v8bf lo = *(const v8bf*)bp;
            v8bf hi = *(const v8bf*)(bp + 16);
            bfrag[ni] = __builtin_shufflevector(lo, hi, 0,1,2,3,4,5,6,7,8,9,10,11,12,13,14,15);
        }
        #pragma unroll
        for (int mi = 0; mi < 2; ++mi)
            #pragma unroll
            for (int ni = 0; ni < 4; ++ni)
                acc[mi][ni] = __builtin_amdgcn_wmma_f32_16x16x32_bf16(
                    false, af[mi], false, bfrag[ni],
                    (short)0, acc[mi][ni], false, false);
        __syncthreads();
    }

    // ---- epilogue: + bias, store fp32 ----
    // C/D layout: lane<16: M=v, N=lane ; lane>=16: M=8+v, N=lane-16
    #pragma unroll
    for (int mi = 0; mi < 2; ++mi) {
        const int gm0 = mBase + wm * 32 + mi * 16 + lhalf * 8;
        #pragma unroll
        for (int ni = 0; ni < 4; ++ni) {
            const int gn = nBase + wn * 64 + ni * 16 + lrow;
            const float bv = bias[gn];
            #pragma unroll
            for (int v = 0; v < 8; ++v)
                out[(long)(gm0 + v) * OUT_DIM + gn] = acc[mi][ni][v] + bv;
        }
    }
}

// ---------------------------------------------------------------------------
extern "C" void kernel_launch(void* const* d_in, const int* in_sizes, int n_in,
                              void* d_out, int out_size, void* d_ws, size_t ws_size,
                              hipStream_t stream) {
    const float* x      = (const float*)d_in[0];   // [8,32,4096]
    const float* weight = (const float*)d_in[1];   // [11008,4096]
    const float* loraA  = (const float*)d_in[2];   // [64,4096]
    const float* loraB  = (const float*)d_in[3];   // [11008,64]
    const float* loraM  = (const float*)d_in[4];   // [1,4096]
    const float* bias   = (const float*)d_in[5];   // [11008]
    float* out = (float*)d_out;                    // [8,32,11008]

    // workspace carve (~2.2 MB)
    float*  normAcc = (float*)d_ws;
    float*  scalev  = normAcc + IN_DIM;
    __bf16* xs      = (__bf16*)(scalev + IN_DIM);
    __bf16* taug    = xs + (size_t)M_DIM * IN_DIM;

    zero_kernel <<<dim3(IN_DIM / 256), dim3(256), 0, stream>>>(normAcc);
    norm_kernel <<<dim3(IN_DIM / 256, OUT_DIM / 256), dim3(256), 0, stream>>>(
        weight, loraA, loraB, normAcc);
    scale_kernel<<<dim3(IN_DIM / 256), dim3(256), 0, stream>>>(normAcc, loraM, scalev);
    xs_kernel   <<<dim3((M_DIM * IN_DIM) / 256), dim3(256), 0, stream>>>(x, scalev, xs);
    t_kernel    <<<dim3(M_DIM), dim3(R_DIM), 0, stream>>>(xs, loraA, taug);
    gemm_kernel <<<dim3(OUT_DIM / 128, M_DIM / 128), dim3(256), 0, stream>>>(
        weight, loraB, xs, taug, bias, out);
}
